// GATNet_601295421670
// MI455X (gfx1250) — matrix operand order repaired and checked by reading, not compiled, not run on hardware
//
#include <hip/hip_runtime.h>
#include <hip/hip_bf16.h>

// ---------------------------------------------------------------------------
// CDNA5 WMMA plumbing
// ---------------------------------------------------------------------------
typedef __attribute__((ext_vector_type(16))) __bf16 v16bf;
typedef __attribute__((ext_vector_type(8)))  float  v8f;

__device__ __forceinline__ v8f wmma_bf16(v16bf a, v16bf b, v8f c) {
  // D = A(16x32 bf16) * B(32x16 bf16) + C(16x16 f32)
  return __builtin_amdgcn_wmma_f32_16x16x32_bf16(false, a, false, b, (short)0, c,
                                                 false, false);
}

__device__ __forceinline__ __bf16 f2bf(float x) {
  unsigned u = __float_as_uint(x);
  unsigned r = (u + 0x7FFFu + ((u >> 16) & 1u)) >> 16;  // round-to-nearest-even
  unsigned short s = (unsigned short)r;
  __bf16 o;
  __builtin_memcpy(&o, &s, 2);
  return o;
}

__device__ __forceinline__ float sigmoidf_(float x) { return 1.f / (1.f + expf(-x)); }

__device__ __forceinline__ void atomicMaxF(float* addr, float val) {
  int* ia = (int*)addr;
  int old = __float_as_int(*addr);
  while (__int_as_float(old) < val) {
    int assumed = old;
    old = atomicCAS(ia, assumed, __float_as_int(val));
    if (old == assumed) break;
  }
}

// ---------------------------------------------------------------------------
// Fragment-swizzled B layout:
//   B (K x N, K%32==0, N%16==0) stored as blocks of 32x16; block (kt,nt) at
//   element offset (kt*(N/16)+nt)*512, holding for lane l (half=l>>4, r=l&15)
//   16 contiguous bf16: element j = B[kt*32 + (j<8 ? half*8+j : 8+half*8+j)][nt*16 + r].
//   A wave fetches one whole WMMA B-fragment as a single 32-byte load.
// ---------------------------------------------------------------------------

// source row-major [K0,N0] (lds = row stride), dst swizzled [Kp,Np]
__global__ void packB_swz(const float* __restrict__ src, __bf16* __restrict__ dst,
                          int K0, int N0, int lds, int Kp, int Np) {
  long i = (long)blockIdx.x * 256 + threadIdx.x;
  if (i >= (long)Kp * Np) return;
  int j = (int)(i & 15);
  long t = i >> 4;
  int lane = (int)(t & 31); t >>= 5;
  int Nt = Np >> 4;
  int nt = (int)(t % Nt), kt = (int)(t / Nt);
  int half = lane >> 4, r = lane & 15;
  int k = kt * 32 + (j < 8 ? half * 8 + j : 16 + half * 8 + (j - 8));
  int n = nt * 16 + r;
  float v = (k < K0 && n < N0) ? src[(long)k * lds + n] : 0.f;
  dst[i] = f2bf(v);
}

// source row-major [N0,K0]; B = src^T  (for torch Wih/Whh: y = x @ W^T)
__global__ void packBT_swz(const float* __restrict__ src, __bf16* __restrict__ dst,
                           int N0, int K0, int Kp, int Np) {
  long i = (long)blockIdx.x * 256 + threadIdx.x;
  if (i >= (long)Kp * Np) return;
  int j = (int)(i & 15);
  long t = i >> 4;
  int lane = (int)(t & 31); t >>= 5;
  int Nt = Np >> 4;
  int nt = (int)(t % Nt), kt = (int)(t / Nt);
  int half = lane >> 4, r = lane & 15;
  int k = kt * 32 + (j < 8 ? half * 8 + j : 16 + half * 8 + (j - 8));
  int n = nt * 16 + r;
  float v = (k < K0 && n < N0) ? src[(long)n * K0 + k] : 0.f;
  dst[i] = f2bf(v);
}

// ---------------------------------------------------------------------------
// Tiled WMMA GEMM with 2x2 register blocking:
//   C[M,Np](f32) = A[M,K](bf16 row-major) * Bw[K,Np](bf16 fragment-swizzled)
//   M%32==0, Np%32==0, K%32==0.  One 32x32 block per wave, 8 waves/block.
//   Per k-step: 2 A frags + 2 B frags -> 4 v_wmma.
// ---------------------------------------------------------------------------
__global__ __launch_bounds__(256)
void wmma_gemm(const __bf16* __restrict__ A, const __bf16* __restrict__ Bw,
               const float* __restrict__ bias, int biasN,
               float* __restrict__ C, int M, int Np, int K, int act) {
  const int lane = threadIdx.x & 31;
  const int wave = threadIdx.x >> 5;
  const int half = lane >> 4, r = lane & 15;
  const long Nb = Np >> 5, Mb = M >> 5;
  const long blk = (long)blockIdx.x * 8 + wave;
  if (blk >= Mb * Nb) return;                       // wave-uniform exit
  const int mb = (int)(blk / Nb), nb = (int)(blk % Nb);
  const int Nt = Np >> 4;
  v8f acc00 = {}, acc01 = {}, acc10 = {}, acc11 = {};
  for (int k = 0; k < K; k += 32) {
    v16bf a0, a1;
    const __bf16* ap0 = A + (long)(mb * 32 + r) * K + k + half * 8;
    const __bf16* ap1 = ap0 + 16L * K;
    __builtin_prefetch(ap0 + 32, 0, 3);             // global_prefetch next K slab
#pragma unroll
    for (int j = 0; j < 8; ++j) {
      a0[j] = ap0[j]; a0[8 + j] = ap0[16 + j];
      a1[j] = ap1[j]; a1[8 + j] = ap1[16 + j];
    }
    const long bbase = ((long)(k >> 5) * Nt + nb * 2) * 512 + lane * 16;
    const v16bf b0 = *(const v16bf*)(Bw + bbase);         // 32B contiguous frag
    const v16bf b1 = *(const v16bf*)(Bw + bbase + 512);
    acc00 = wmma_bf16(a0, b0, acc00);
    acc01 = wmma_bf16(a0, b1, acc01);
    acc10 = wmma_bf16(a1, b0, acc10);
    acc11 = wmma_bf16(a1, b1, acc11);
  }
#pragma unroll
  for (int q = 0; q < 4; ++q) {
    const v8f acc = q == 0 ? acc00 : q == 1 ? acc01 : q == 2 ? acc10 : acc11;
    const int row0 = mb * 32 + (q >> 1) * 16;
    const int col = nb * 32 + (q & 1) * 16 + r;
    const float bv = (bias && col < biasN) ? bias[col] : 0.f;
#pragma unroll
    for (int i = 0; i < 8; ++i) {
      float v = acc[i] + bv;
      if (act == 1) v = fmaxf(v, 0.f);
      C[(long)(row0 + i + 8 * half) * Np + col] = v;
    }
  }
}

// ---------------------------------------------------------------------------
// Packing / elementwise helpers
// ---------------------------------------------------------------------------
__global__ void fill_f32(float* p, float v, long n) {
  long i = (long)blockIdx.x * 256 + threadIdx.x;
  if (i < n) p[i] = v;
}

// row-major A-operand pack: dst[r*ldd+c] = (r<R0 && c<C0) ? bf16(src[r*lds+c]) : 0
__global__ void pack_bf16(const float* __restrict__ src, __bf16* __restrict__ dst,
                          int R0, int C0, int lds, int Rp, int ldd) {
  long i = (long)blockIdx.x * 256 + threadIdx.x;
  if (i >= (long)Rp * ldd) return;
  int r = (int)(i / ldd), c = (int)(i % ldd);
  float v = (r < R0 && c < C0) ? src[(long)r * lds + c] : 0.f;
  dst[i] = f2bf(v);
}

__global__ void bias_act(float* data, const float* __restrict__ b, long R, int C,
                         int ld, int act) {
  long i = (long)blockIdx.x * 256 + threadIdx.x;
  if (i >= R * C) return;
  long r = i / C; int c = (int)(i % C);
  float v = data[r * ld + c] + b[c];
  if (act == 1) v = fmaxf(v, 0.f);
  else if (act == 2) v = v > 0.f ? v : expm1f(v);
  data[r * ld + c] = v;
}

// ---------------------------------------------------------------------------
// GAT edge pipeline
// ---------------------------------------------------------------------------
__device__ __forceinline__ void edge_sd(const int* ei, int E, long e, int& s, int& d) {
  if (e < E) { s = ei[e]; d = ei[(long)E + e]; }
  else       { int n = (int)(e - E); s = n; d = n; }   // self loops
}

__global__ void attn_coef(const float* __restrict__ hmat, const float* __restrict__ att,
                          float* __restrict__ out, int N, int H, int OC, int ld) {
  long i = (long)blockIdx.x * 256 + threadIdx.x;
  if (i >= (long)N * H) return;
  int n = (int)(i / H), h = (int)(i % H);
  const float* hp = hmat + (long)n * ld + h * OC;
  const float* ap = att + h * OC;
  float s = 0.f;
  for (int c = 0; c < OC; ++c) s += hp[c] * ap[c];
  out[i] = s;
}

__global__ void edge_max_k(const int* ei, int E, int EE, const float* as, const float* ad,
                           float* emax, int H) {
  long i = (long)blockIdx.x * 256 + threadIdx.x;
  if (i >= (long)EE * H) return;
  long e = i / H; int h = (int)(i % H);
  int s, d; edge_sd(ei, E, e, s, d);
  float v = as[(long)s * H + h] + ad[(long)d * H + h];
  v = v > 0.f ? v : 0.2f * v;
  atomicMaxF(&emax[(long)d * H + h], v);
}

__global__ void edge_den_k(const int* ei, int E, int EE, const float* as, const float* ad,
                           const float* emax, float* den, int H) {
  long i = (long)blockIdx.x * 256 + threadIdx.x;
  if (i >= (long)EE * H) return;
  long e = i / H; int h = (int)(i % H);
  int s, d; edge_sd(ei, E, e, s, d);
  float v = as[(long)s * H + h] + ad[(long)d * H + h];
  v = v > 0.f ? v : 0.2f * v;
  atomicAdd(&den[(long)d * H + h], expf(v - emax[(long)d * H + h]));
}

__global__ void edge_aggr_k(const int* ei, int E, int EE, const float* as, const float* ad,
                            const float* emax, const float* den,
                            const float* __restrict__ hmat, float* __restrict__ o,
                            int H, int OC, int ld) {
  long i = (long)blockIdx.x * 256 + threadIdx.x;
  if (i >= (long)EE * H) return;
  long e = i / H; int h = (int)(i % H);
  int s, d; edge_sd(ei, E, e, s, d);
  float v = as[(long)s * H + h] + ad[(long)d * H + h];
  v = v > 0.f ? v : 0.2f * v;
  float alpha = expf(v - emax[(long)d * H + h]) / (den[(long)d * H + h] + 1e-16f);
  const float* hp = hmat + (long)s * ld + h * OC;
  float* op = o + (long)d * ld + h * OC;
  for (int c = 0; c < OC; ++c) atomicAdd(&op[c], hp[c] * alpha);
}

__global__ void pool_max(const float* __restrict__ h, const int* __restrict__ batch,
                         float* g, int N, int C) {
  long i = (long)blockIdx.x * 256 + threadIdx.x;
  if (i >= (long)N * C) return;
  int n = (int)(i / C), c = (int)(i % C);
  atomicMaxF(&g[(long)batch[n] * C + c], h[(long)n * C + c]);
}

// ---------------------------------------------------------------------------
// Embedding gathers (A-matrix packing for the big GI GEMMs)
// ---------------------------------------------------------------------------
__global__ void gather_pack_emb(const int* __restrict__ idxs, const float* __restrict__ emb,
                                __bf16* __restrict__ dst, long M, int C, int ld) {
  long i = (long)blockIdx.x * 256 + threadIdx.x;
  if (i >= M * ld) return;
  long m = i / ld; int c = (int)(i % ld);
  float v = (c < C) ? emb[(long)idxs[m] * C + c] : 0.f;
  dst[i] = f2bf(v);
}

// xs[t,b,:] = embed_xt[target[b,t]]  -> row m = t*B + b
__global__ void gather_pack_target(const int* __restrict__ target, const float* __restrict__ emb,
                                   __bf16* __restrict__ dst, int T, int B, int C) {
  long i = (long)blockIdx.x * 256 + threadIdx.x;
  if (i >= (long)T * B * C) return;
  long m = i / C; int c = (int)(i % C);
  int t = (int)(m / B), b = (int)(m % B);
  dst[i] = f2bf(emb[(long)target[(long)b * T + t] * C + c]);
}

__global__ void concat_pack(const float* __restrict__ yf, const float* __restrict__ yb,
                            __bf16* __restrict__ dst, long M) {
  long i = (long)blockIdx.x * 256 + threadIdx.x;
  if (i >= M * 64) return;
  long m = i / 64; int d = (int)(i % 64);
  float v = d < 32 ? yf[m * 32 + d] : yb[m * 32 + (d - 32)];
  dst[i] = f2bf(v);
}

// ---------------------------------------------------------------------------
// Persistent bidirectional GRU (SMILES branch).
// GI[t,b,0:300] precomputed (= x@Wih^T + bih).  Per step: gh = h @ Whh^T via
// WMMA with LDS-resident bf16 state (A) and swizzled L2-hot weights (B).
// Dims: steps=128, batch=200 (pad 208), hidden 100 (pad 128), gates 300 (pad 320)
// ---------------------------------------------------------------------------
__global__ __launch_bounds__(1024)
void gru_persistent(const float* __restrict__ GIf, const float* __restrict__ GIb,
                    const __bf16* __restrict__ WTf, const __bf16* __restrict__ WTb,
                    const float* __restrict__ bhhf, const float* __restrict__ bhhb,
                    float* __restrict__ ghf, float* __restrict__ ghb,
                    float* __restrict__ hff, float* __restrict__ hfb,
                    float* __restrict__ acc_out) {
  const int dir = blockIdx.x;
  const float* GI = dir ? GIb : GIf;
  const __bf16* WT = dir ? WTb : WTf;
  const float* bhh = dir ? bhhb : bhhf;
  float* gh = dir ? ghb : ghf;
  float* hF = dir ? hfb : hff;
  const int tid = threadIdx.x;
  const int wave = tid >> 5, lane = tid & 31, half = lane >> 4, r = lane & 15;
  __shared__ __align__(32) __bf16 hS[208 * 128];        // 53 KB bf16 state
  for (int i = tid; i < 208 * 128; i += 1024) hS[i] = f2bf(0.f);
  for (int i = tid; i < 200 * 100; i += 1024) hF[i] = 0.f;
  __syncthreads();
  for (int s = 0; s < 128; ++s) {
    const int t = dir ? (127 - s) : s;
    // phase 1: gh[208x320] = hS[208x128] @ WT[128x320]  (WMMA)
    for (int tile = wave; tile < 13 * 20; tile += 32) {
      const int mt = tile / 20, nt = tile % 20;
      v8f acc = {};
      for (int k = 0; k < 128; k += 32) {
        v16bf a;
        const __bf16* ap = &hS[(mt * 16 + r) * 128 + k + half * 8];
#pragma unroll
        for (int j = 0; j < 8; ++j) { a[j] = ap[j]; a[8 + j] = ap[16 + j]; }
        const v16bf b = *(const v16bf*)(WT + ((long)(k >> 5) * 20 + nt) * 512 + lane * 16);
        acc = wmma_bf16(a, b, acc);
      }
      const int col = nt * 16 + r;
#pragma unroll
      for (int j = 0; j < 8; ++j) gh[(mt * 16 + j + 8 * half) * 320 + col] = acc[j];
    }
    __syncthreads();
    // phase 2: gates + state update + relu-mean accumulation
    for (int i = tid; i < 200 * 100; i += 1024) {
      const int bb = i / 100, j = i % 100;
      const long gib = ((long)t * 200 + bb) * 320;
      const float gr = GI[gib + j], gz = GI[gib + 100 + j], gn = GI[gib + 200 + j];
      const float hr = gh[bb * 320 + j] + bhh[j];
      const float hz = gh[bb * 320 + 100 + j] + bhh[100 + j];
      const float hn = gh[bb * 320 + 200 + j] + bhh[200 + j];
      const float rg = sigmoidf_(gr + hr);
      const float zg = sigmoidf_(gz + hz);
      const float ng = tanhf(gn + rg * hn);
      float h = hF[i];
      h = (1.f - zg) * ng + zg * h;
      hF[i] = h;
      hS[bb * 128 + j] = f2bf(h);
      atomicAdd(&acc_out[t * 200 + dir * 100 + j], fmaxf(h, 0.f) * 0.005f);  // /200
    }
    __syncthreads();
  }
}

// ---------------------------------------------------------------------------
// Persistent bidirectional LSTM layer (protein branch).
// Dims: T steps, batch 128, hidden 32, gates 128.  All state in LDS (48 KB);
// weights LDS-resident in fragment-swizzled layout -> two ds_load_b128/frag.
// ---------------------------------------------------------------------------
__global__ __launch_bounds__(1024)
void lstm_persistent(const float* __restrict__ GIf, const float* __restrict__ GIb,
                     const __bf16* __restrict__ WTf, const __bf16* __restrict__ WTb,
                     const float* __restrict__ bhhf, const float* __restrict__ bhhb,
                     float* __restrict__ ghf_, float* __restrict__ ghb_,
                     float* __restrict__ yf, float* __restrict__ yb,
                     float* __restrict__ hTf, float* __restrict__ hTb, int T) {
  const int dir = blockIdx.x;
  const float* GI = dir ? GIb : GIf;
  const __bf16* WT = dir ? WTb : WTf;
  const float* bhh = dir ? bhhb : bhhf;
  float* gh = dir ? ghb_ : ghf_;
  float* y = dir ? yb : yf;
  float* hT = dir ? hTb : hTf;
  const int tid = threadIdx.x;
  const int wave = tid >> 5, lane = tid & 31, half = lane >> 4, r = lane & 15;
  __shared__ __align__(32) __bf16 whS[32 * 128];   // swizzled (kt=0, nt=0..7)
  __shared__ __align__(32) __bf16 hS[128 * 32];
  __shared__ float hF[128 * 32];
  __shared__ float cF[128 * 32];
  for (int i = tid; i < 32 * 128; i += 1024) whS[i] = WT[i];   // already swizzled
  for (int i = tid; i < 128 * 32; i += 1024) { hS[i] = f2bf(0.f); hF[i] = 0.f; cF[i] = 0.f; }
  __syncthreads();
  for (int s = 0; s < T; ++s) {
    const int t = dir ? (T - 1 - s) : s;
    // phase 1: gh[128x128] = hS[128x32] @ whS[32x128]  (single-K WMMA per tile)
    for (int tile = wave; tile < 64; tile += 32) {
      const int mt = tile >> 3, nt = tile & 7;
      v16bf a;
      const __bf16* ap = &hS[(mt * 16 + r) * 32 + half * 8];
#pragma unroll
      for (int j = 0; j < 8; ++j) { a[j] = ap[j]; a[8 + j] = ap[16 + j]; }
      const v16bf b = *(const v16bf*)&whS[(nt * 32 + lane) << 4];  // ds_load_b128x2
      v8f acc = {};
      acc = wmma_bf16(a, b, acc);
      const int col = nt * 16 + r;
#pragma unroll
      for (int j = 0; j < 8; ++j) gh[(mt * 16 + j + 8 * half) * 128 + col] = acc[j];
    }
    __syncthreads();
    // phase 2: LSTM cell
    for (int i = tid; i < 128 * 32; i += 1024) {
      const int bb = i >> 5, u = i & 31;
      const long gib = ((long)t * 128 + bb) * 128;
      const float gi = GI[gib + u] + gh[bb * 128 + u] + bhh[u];
      const float gf = GI[gib + 32 + u] + gh[bb * 128 + 32 + u] + bhh[32 + u];
      const float gg = GI[gib + 64 + u] + gh[bb * 128 + 64 + u] + bhh[64 + u];
      const float go = GI[gib + 96 + u] + gh[bb * 128 + 96 + u] + bhh[96 + u];
      float c = sigmoidf_(gf) * cF[i] + sigmoidf_(gi) * tanhf(gg);
      float h = sigmoidf_(go) * tanhf(c);
      cF[i] = c; hF[i] = h;
      hS[i] = f2bf(h);
      y[(long)t * 4096 + i] = h;
    }
    __syncthreads();
  }
  for (int i = tid; i < 4096; i += 1024) hT[i] = hF[i];
}

// ---------------------------------------------------------------------------
// Attention over protein LSTM outputs
// ---------------------------------------------------------------------------
__global__ void attn_logits_k(const float* __restrict__ yf, const float* __restrict__ yb,
                              const float* __restrict__ hidden, float* __restrict__ logits,
                              int B, int T) {
  long i = (long)blockIdx.x * 256 + threadIdx.x;
  if (i >= (long)B * T * 3) return;
  int k = (int)(i % 3);
  long bt = i / 3;
  int t = (int)(bt % T), b = (int)(bt / T);
  const long m = (long)t * B + b;
  const float* hb = hidden + (long)b * 192;
  float s = 0.f;
  for (int d = 0; d < 32; ++d) s += yf[m * 32 + d] * hb[d * 3 + k];
  for (int d = 0; d < 32; ++d) s += yb[m * 32 + d] * hb[(32 + d) * 3 + k];
  logits[i] = s;  // layout [b,t,k]
}

__device__ float blk_reduce_max(float v, float* red) {
  int tid = threadIdx.x;
  red[tid] = v; __syncthreads();
  for (int s = 128; s > 0; s >>= 1) {
    if (tid < s) red[tid] = fmaxf(red[tid], red[tid + s]);
    __syncthreads();
  }
  float r = red[0]; __syncthreads();
  return r;
}
__device__ float blk_reduce_sum(float v, float* red) {
  int tid = threadIdx.x;
  red[tid] = v; __syncthreads();
  for (int s = 128; s > 0; s >>= 1) {
    if (tid < s) red[tid] += red[tid + s];
    __syncthreads();
  }
  float r = red[0]; __syncthreads();
  return r;
}

__global__ __launch_bounds__(256)
void attn_context(const float* __restrict__ yf, const float* __restrict__ yb,
                  const float* __restrict__ logits, float* __restrict__ xt, int B, int T) {
  const int b = blockIdx.x, tid = threadIdx.x;
  __shared__ float red[256];
  __shared__ float mx[3], sm[3];
  for (int k = 0; k < 3; ++k) {
    float m = -1e30f;
    for (int t = tid; t < T; t += 256) m = fmaxf(m, logits[((long)b * T + t) * 3 + k]);
    m = blk_reduce_max(m, red);
    if (tid == 0) mx[k] = m;
    __syncthreads();
    float s = 0.f;
    for (int t = tid; t < T; t += 256) s += expf(logits[((long)b * T + t) * 3 + k] - mx[k]);
    s = blk_reduce_sum(s, red);
    if (tid == 0) sm[k] = s;
    __syncthreads();
  }
  if (tid < 192) {
    const int d = tid / 3, k = tid % 3;
    float acc = 0.f;
    for (int t = 0; t < T; ++t) {
      const long m = (long)t * B + b;
      const float v = d < 32 ? yf[m * 32 + d] : yb[m * 32 + (d - 32)];
      const float w = expf(logits[((long)b * T + t) * 3 + k] - mx[k]) / sm[k];
      acc += v * w;
    }
    xt[b * 192 + tid] = acc;  // context[b].reshape(-1) over (d,k)
  }
}

// ---------------------------------------------------------------------------
// Final concat + output copy
// ---------------------------------------------------------------------------
__global__ void build_xc(const float* __restrict__ g1, const float* __restrict__ asml,
                         const float* __restrict__ xt, __bf16* __restrict__ dst, int B) {
  long i = (long)blockIdx.x * 256 + threadIdx.x;
  if (i >= (long)B * 544) return;
  int b = (int)(i / 544), c = (int)(i % 544);
  float v = 0.f;
  if (c < 128)      v = g1[b * 128 + c];
  else if (c < 328) v = asml[b * 200 + (c - 128)];
  else if (c < 520) v = xt[b * 192 + (c - 328)];
  dst[i] = f2bf(v);
}

__global__ void final_copy(const float* __restrict__ o, float* __restrict__ out, int B) {
  int b = blockIdx.x * 256 + threadIdx.x;
  if (b < B) out[b] = o[b * 32];
}

// ---------------------------------------------------------------------------
// Host orchestration
// ---------------------------------------------------------------------------
static inline int nblk(long total) { return (int)((total + 255) / 256); }

static void launch_gemm(const __bf16* A, const __bf16* B, const float* bias, int biasN,
                        float* C, int M, int Np, int K, int act, hipStream_t s) {
  long blks = ((long)M / 32) * (Np / 32);           // 32x32 per wave
  wmma_gemm<<<(int)((blks + 7) / 8), 256, 0, s>>>(A, B, bias, biasN, C, M, Np, K, act);
}

extern "C" void kernel_launch(void* const* d_in, const int* in_sizes, int n_in,
                              void* d_out, int out_size, void* d_ws, size_t ws_size,
                              hipStream_t stream) {
  (void)n_in; (void)out_size; (void)ws_size;
  // ---- inputs -----------------------------------------------------------
  const float* x   = (const float*)d_in[0];
  const int* ei    = (const int*)d_in[1];
  const int* batch = (const int*)d_in[2];
  const int* smile = (const int*)d_in[3];
  const int* targ  = (const int*)d_in[4];
  auto F = [&](int i) { return (const float*)d_in[i]; };
  // params (flattened in setup_inputs dict order)
  const float *g1W = F(5), *g1as = F(6), *g1ad = F(7), *g1b = F(8);
  const float *g2W = F(9), *g2as = F(10), *g2ad = F(11), *g2b = F(12);
  const float *fgW = F(13), *fgB = F(14);
  const float *embS = F(15);
  const float *gWihF = F(16), *gWhhF = F(17), *gBihF = F(18), *gBhhF = F(19);
  const float *gWihB = F(20), *gWhhB = F(21), *gBihB = F(22), *gBhhB = F(23);
  const float *embT = F(24);
  const float *lWih[3][2], *lWhh[3][2], *lBih[3][2], *lBhh[3][2];
  for (int l = 0; l < 3; ++l) {
    int base = 25 + l * 8;
    lWih[l][0] = F(base + 0); lWhh[l][0] = F(base + 1);
    lBih[l][0] = F(base + 2); lBhh[l][0] = F(base + 3);
    lWih[l][1] = F(base + 4); lWhh[l][1] = F(base + 5);
    lBih[l][1] = F(base + 6); lBhh[l][1] = F(base + 7);
  }
  const float *fc1W = F(49), *fc1B = F(50), *fc2W = F(51), *fc2B = F(52);
  const float *outW = F(53), *outB = F(54);

  const int N = in_sizes[0] / 78;       // 16000
  const int E = in_sizes[1] / 2;        // 160000
  const int B = in_sizes[3] / 200;      // 128
  const int T = in_sizes[4] / B;        // 1000
  const int EE = E + N;
  const long GM = (long)T * B;          // 128000 LSTM rows
  const long SM = 128L * 200;           // 25600 GRU rows

  // ---- workspace carve --------------------------------------------------
  char* wp = (char*)d_ws;
  auto alloc = [&](size_t bytes) -> void* {
    void* p = wp; wp += (bytes + 255) & ~(size_t)255; return p;
  };
  __bf16* xp   = (__bf16*)alloc((size_t)N * 96 * 2);
  __bf16* W1p  = (__bf16*)alloc(96 * 800 * 2);
  float*  h1   = (float*)alloc((size_t)N * 800 * 4);
  float*  as1  = (float*)alloc((size_t)N * 10 * 4);
  float*  ad1  = (float*)alloc((size_t)N * 10 * 4);
  float*  em1  = (float*)alloc((size_t)N * 10 * 4);
  float*  dn1  = (float*)alloc((size_t)N * 10 * 4);
  float*  o1   = (float*)alloc((size_t)N * 800 * 4);
  __bf16* h1b  = (__bf16*)alloc((size_t)N * 800 * 2);
  __bf16* W2p  = (__bf16*)alloc(800 * 128 * 2);
  float*  h2   = (float*)alloc((size_t)N * 128 * 4);
  float*  as2  = (float*)alloc((size_t)N * 4);
  float*  ad2  = (float*)alloc((size_t)N * 4);
  float*  em2  = (float*)alloc((size_t)N * 4);
  float*  dn2  = (float*)alloc((size_t)N * 4);
  float*  o2   = (float*)alloc((size_t)N * 128 * 4);
  float*  g    = (float*)alloc((size_t)B * 128 * 4);
  __bf16* gb   = (__bf16*)alloc((size_t)B * 128 * 2);
  __bf16* Wgp  = (__bf16*)alloc(128 * 128 * 2);
  float*  g1o  = (float*)alloc((size_t)B * 128 * 4);
  // GRU
  __bf16* svp   = (__bf16*)alloc((size_t)SM * 128 * 2);
  __bf16* WihTf = (__bf16*)alloc(128 * 320 * 2);
  __bf16* WihTb = (__bf16*)alloc(128 * 320 * 2);
  __bf16* WhhTf = (__bf16*)alloc(128 * 320 * 2);
  __bf16* WhhTb = (__bf16*)alloc(128 * 320 * 2);
  float*  GIfG  = (float*)alloc((size_t)SM * 320 * 4);
  float*  GIbG  = (float*)alloc((size_t)SM * 320 * 4);
  float*  ghfG  = (float*)alloc(208 * 320 * 4);
  float*  ghbG  = (float*)alloc(208 * 320 * 4);
  float*  hfG   = (float*)alloc(200 * 100 * 4);
  float*  hbG   = (float*)alloc(200 * 100 * 4);
  float*  asml  = (float*)alloc(128 * 200 * 4);
  // LSTM
  __bf16* xtp   = (__bf16*)alloc((size_t)GM * 128 * 2);
  __bf16* catp  = (__bf16*)alloc((size_t)GM * 64 * 2);
  __bf16* LWihT[3][2]; __bf16* LWhhT[3][2];
  for (int l = 0; l < 3; ++l)
    for (int d = 0; d < 2; ++d) {
      LWihT[l][d] = (__bf16*)alloc(128 * 128 * 2);
      LWhhT[l][d] = (__bf16*)alloc(32 * 128 * 2);
    }
  float* GIfL = (float*)alloc((size_t)GM * 128 * 4);
  float* GIbL = (float*)alloc((size_t)GM * 128 * 4);
  float* ghfL = (float*)alloc(128 * 128 * 4);
  float* ghbL = (float*)alloc(128 * 128 * 4);
  float* yfL  = (float*)alloc((size_t)GM * 32 * 4);
  float* ybL  = (float*)alloc((size_t)GM * 32 * 4);
  float* hn   = (float*)alloc(6 * 128 * 32 * 4);   // == hidden[B,64,3] flat
  float* lgts = (float*)alloc((size_t)B * T * 3 * 4);
  float* xt   = (float*)alloc((size_t)B * 192 * 4);
  // FC head
  __bf16* xcb   = (__bf16*)alloc((size_t)B * 544 * 2);
  __bf16* fc1Wp = (__bf16*)alloc(544 * 1024 * 2);
  float*  ofc1  = (float*)alloc((size_t)B * 1024 * 4);
  __bf16* b1    = (__bf16*)alloc((size_t)B * 1024 * 2);
  __bf16* fc2Wp = (__bf16*)alloc((size_t)1024 * 256 * 2);
  float*  ofc2  = (float*)alloc((size_t)B * 256 * 4);
  __bf16* b2    = (__bf16*)alloc((size_t)B * 256 * 2);
  __bf16* Wop   = (__bf16*)alloc(256 * 32 * 2);
  float*  oout  = (float*)alloc((size_t)B * 32 * 4);

  // ---- GAT layer 1 ------------------------------------------------------
  pack_bf16<<<nblk((long)N * 96), 256, 0, stream>>>(x, xp, N, 78, 78, N, 96);
  packB_swz<<<nblk(96L * 800), 256, 0, stream>>>(g1W, W1p, 78, 780, 780, 96, 800);
  launch_gemm(xp, W1p, nullptr, 0, h1, N, 800, 96, 0, stream);
  attn_coef<<<nblk((long)N * 10), 256, 0, stream>>>(h1, g1as, as1, N, 10, 78, 800);
  attn_coef<<<nblk((long)N * 10), 256, 0, stream>>>(h1, g1ad, ad1, N, 10, 78, 800);
  fill_f32<<<nblk((long)N * 10), 256, 0, stream>>>(em1, -1e30f, (long)N * 10);
  fill_f32<<<nblk((long)N * 10), 256, 0, stream>>>(dn1, 0.f, (long)N * 10);
  fill_f32<<<nblk((long)N * 800), 256, 0, stream>>>(o1, 0.f, (long)N * 800);
  edge_max_k<<<nblk((long)EE * 10), 256, 0, stream>>>(ei, E, EE, as1, ad1, em1, 10);
  edge_den_k<<<nblk((long)EE * 10), 256, 0, stream>>>(ei, E, EE, as1, ad1, em1, dn1, 10);
  edge_aggr_k<<<nblk((long)EE * 10), 256, 0, stream>>>(ei, E, EE, as1, ad1, em1, dn1,
                                                       h1, o1, 10, 78, 800);
  bias_act<<<nblk((long)N * 780), 256, 0, stream>>>(o1, g1b, N, 780, 800, 2);  // ELU
  pack_bf16<<<nblk((long)N * 800), 256, 0, stream>>>(o1, h1b, N, 780, 800, N, 800);

  // ---- GAT layer 2 + pool + fc_g1 --------------------------------------
  packB_swz<<<nblk(800L * 128), 256, 0, stream>>>(g2W, W2p, 780, 128, 128, 800, 128);
  launch_gemm(h1b, W2p, nullptr, 0, h2, N, 128, 800, 0, stream);
  attn_coef<<<nblk((long)N), 256, 0, stream>>>(h2, g2as, as2, N, 1, 128, 128);
  attn_coef<<<nblk((long)N), 256, 0, stream>>>(h2, g2ad, ad2, N, 1, 128, 128);
  fill_f32<<<nblk((long)N), 256, 0, stream>>>(em2, -1e30f, N);
  fill_f32<<<nblk((long)N), 256, 0, stream>>>(dn2, 0.f, N);
  fill_f32<<<nblk((long)N * 128), 256, 0, stream>>>(o2, 0.f, (long)N * 128);
  edge_max_k<<<nblk((long)EE), 256, 0, stream>>>(ei, E, EE, as2, ad2, em2, 1);
  edge_den_k<<<nblk((long)EE), 256, 0, stream>>>(ei, E, EE, as2, ad2, em2, dn2, 1);
  edge_aggr_k<<<nblk((long)EE), 256, 0, stream>>>(ei, E, EE, as2, ad2, em2, dn2,
                                                  h2, o2, 1, 128, 128);
  bias_act<<<nblk((long)N * 128), 256, 0, stream>>>(o2, g2b, N, 128, 128, 1);  // ReLU
  fill_f32<<<nblk((long)B * 128), 256, 0, stream>>>(g, -1e30f, (long)B * 128);
  pool_max<<<nblk((long)N * 128), 256, 0, stream>>>(o2, batch, g, N, 128);
  pack_bf16<<<nblk((long)B * 128), 256, 0, stream>>>(g, gb, B, 128, 128, B, 128);
  packB_swz<<<nblk(128L * 128), 256, 0, stream>>>(fgW, Wgp, 128, 128, 128, 128, 128);
  launch_gemm(gb, Wgp, fgB, 128, g1o, B, 128, 128, 1, stream);

  // ---- SMILES GRU branch ------------------------------------------------
  gather_pack_emb<<<nblk(SM * 128), 256, 0, stream>>>(smile, embS, svp, SM, 100, 128);
  packBT_swz<<<nblk(128L * 320), 256, 0, stream>>>(gWihF, WihTf, 300, 100, 128, 320);
  packBT_swz<<<nblk(128L * 320), 256, 0, stream>>>(gWihB, WihTb, 300, 100, 128, 320);
  packBT_swz<<<nblk(128L * 320), 256, 0, stream>>>(gWhhF, WhhTf, 300, 100, 128, 320);
  packBT_swz<<<nblk(128L * 320), 256, 0, stream>>>(gWhhB, WhhTb, 300, 100, 128, 320);
  launch_gemm(svp, WihTf, gBihF, 300, GIfG, (int)SM, 320, 128, 0, stream);
  launch_gemm(svp, WihTb, gBihB, 300, GIbG, (int)SM, 320, 128, 0, stream);
  fill_f32<<<nblk(128L * 200), 256, 0, stream>>>(asml, 0.f, 128L * 200);
  gru_persistent<<<2, 1024, 0, stream>>>(GIfG, GIbG, WhhTf, WhhTb, gBhhF, gBhhB,
                                         ghfG, ghbG, hfG, hbG, asml);

  // ---- protein LSTM branch (3 bidirectional layers) ---------------------
  gather_pack_target<<<nblk(GM * 128), 256, 0, stream>>>(targ, embT, xtp, T, B, 128);
  for (int l = 0; l < 3; ++l)
    for (int d = 0; d < 2; ++d) {
      int K0 = (l == 0) ? 128 : 64;
      packBT_swz<<<nblk((long)K0 * 128), 256, 0, stream>>>(lWih[l][d], LWihT[l][d],
                                                           128, K0, K0, 128);
      packBT_swz<<<nblk(32L * 128), 256, 0, stream>>>(lWhh[l][d], LWhhT[l][d],
                                                      128, 32, 32, 128);
    }
  for (int l = 0; l < 3; ++l) {
    if (l == 0) {
      launch_gemm(xtp, LWihT[0][0], lBih[0][0], 128, GIfL, (int)GM, 128, 128, 0, stream);
      launch_gemm(xtp, LWihT[0][1], lBih[0][1], 128, GIbL, (int)GM, 128, 128, 0, stream);
    } else {
      concat_pack<<<nblk(GM * 64), 256, 0, stream>>>(yfL, ybL, catp, GM);
      launch_gemm(catp, LWihT[l][0], lBih[l][0], 128, GIfL, (int)GM, 128, 64, 0, stream);
      launch_gemm(catp, LWihT[l][1], lBih[l][1], 128, GIbL, (int)GM, 128, 64, 0, stream);
    }
    lstm_persistent<<<2, 1024, 0, stream>>>(GIfL, GIbL, LWhhT[l][0], LWhhT[l][1],
                                            lBhh[l][0], lBhh[l][1], ghfL, ghbL,
                                            yfL, ybL,
                                            hn + (l * 2 + 0) * 4096,
                                            hn + (l * 2 + 1) * 4096, T);
  }
  // attention (hn flat [6,128,32] == hidden [B,64,3] flat after torch .view)
  attn_logits_k<<<nblk((long)B * T * 3), 256, 0, stream>>>(yfL, ybL, hn, lgts, B, T);
  attn_context<<<B, 256, 0, stream>>>(yfL, ybL, lgts, xt, B, T);

  // ---- FC head ----------------------------------------------------------
  build_xc<<<nblk((long)B * 544), 256, 0, stream>>>(g1o, asml, xt, xcb, B);
  packB_swz<<<nblk(544L * 1024), 256, 0, stream>>>(fc1W, fc1Wp, 520, 1024, 1024, 544, 1024);
  launch_gemm(xcb, fc1Wp, fc1B, 1024, ofc1, B, 1024, 544, 1, stream);
  pack_bf16<<<nblk((long)B * 1024), 256, 0, stream>>>(ofc1, b1, B, 1024, 1024, B, 1024);
  packB_swz<<<nblk(1024L * 256), 256, 0, stream>>>(fc2W, fc2Wp, 1024, 256, 256, 1024, 256);
  launch_gemm(b1, fc2Wp, fc2B, 256, ofc2, B, 256, 1024, 1, stream);
  pack_bf16<<<nblk((long)B * 256), 256, 0, stream>>>(ofc2, b2, B, 256, 256, B, 256);
  packB_swz<<<nblk(256L * 32), 256, 0, stream>>>(outW, Wop, 256, 1, 1, 256, 32);
  launch_gemm(b2, Wop, outB, 1, oout, B, 32, 256, 0, stream);
  final_copy<<<nblk(B), 256, 0, stream>>>(oout, (float*)d_out, B);
}
